// Partial_conv3_24412594110555
// MI455X (gfx1250) — matrix operand order repaired and checked by reading, not compiled, not run on hardware
//
#include <hip/hip_runtime.h>

// ---------------------------------------------------------------------------
// x: (4,128,64,64) f32. Token selection is analytically the identity split
// (softmax rows sum to 1 -> token_w == 1/4096; top_k ties pick lowest index):
// X1 = first 2048 elems of each HW plane viewed (64,32), X2 = second 2048.
// ---------------------------------------------------------------------------
#define Bn   4
#define Cc   128
#define H1   64
#define W1   32
#define HW1  (H1*W1)            // 2048
#define KK   9
#define CK   (Cc*KK)            // 1152 = GEMM K dim
#define OC   128

typedef __attribute__((ext_vector_type(16))) __bf16 v16bf;
typedef __attribute__((ext_vector_type(8)))  __bf16 v8bf;
typedef __attribute__((ext_vector_type(8)))  float  v8f;

static __device__ __forceinline__ int iclamp(int v, int lo, int hi) {
    return v < lo ? lo : (v > hi ? hi : v);
}

static __device__ __forceinline__ v16bf cat16(v8bf lo, v8bf hi) {
    return __builtin_shufflevector(lo, hi, 0, 1, 2, 3, 4, 5, 6, 7,
                                           8, 9, 10, 11, 12, 13, 14, 15);
}

// ---------------------------------------------------------------------------
// K0: f32 -> bf16 weight conversion ((128 x 1152) flat)
// ---------------------------------------------------------------------------
__global__ __launch_bounds__(256)
void cvt_bf16_kernel(const float* __restrict__ in, __bf16* __restrict__ out, int n) {
    int i = blockIdx.x * blockDim.x + threadIdx.x;
    if (i < n) out[i] = (__bf16)in[i];
}

// ---------------------------------------------------------------------------
// K1: offset conv: 3x3, pad 1, on X1 view (Bn,Cc,64,32) -> (Bn,18,64,32) f32
// ---------------------------------------------------------------------------
__global__ __launch_bounds__(256)
void offset_conv_kernel(const float* __restrict__ x,
                        const float* __restrict__ w,     // (18,128,3,3)
                        const float* __restrict__ bias,  // (18,)
                        float* __restrict__ off)         // (Bn,18,64,32)
{
    int idx = blockIdx.x * blockDim.x + threadIdx.x;     // 147456
    if (idx >= Bn * 18 * HW1) return;
    int j = idx & 31;
    int i = (idx >> 5) & 63;
    int k = (idx >> 11) % 18;
    int b = idx / (18 * HW1);

    const float* xb = x + (size_t)b * Cc * 4096;         // X1 = first half of plane
    float acc = bias[k];
    for (int c = 0; c < Cc; ++c) {
        const float* plane = xb + (size_t)c * 4096;      // (64,32) view
        const float* wk = w + ((size_t)k * Cc + c) * 9;
#pragma unroll
        for (int kh = 0; kh < 3; ++kh) {
            int ii = i + kh - 1;
            if (ii < 0 || ii >= H1) continue;
#pragma unroll
            for (int kw = 0; kw < 3; ++kw) {
                int jj = j + kw - 1;
                if (jj < 0 || jj >= W1) continue;
                acc += plane[ii * W1 + jj] * wk[kh * 3 + kw];
            }
        }
    }
    off[idx] = acc;
}

// ---------------------------------------------------------------------------
// K2a: deformable bilinear sampling -> K-major im2col: BmT[b][n][c*9+k], bf16.
// One thread per (b,k,i,j); bilinear weights shared across all 128 channels.
// ---------------------------------------------------------------------------
__global__ __launch_bounds__(256)
void deform_sample_kernel(const float* __restrict__ x,
                          const float* __restrict__ off,   // (Bn,18,64,32)
                          __bf16* __restrict__ BmT)        // (Bn,2048,1152)
{
    int idx = blockIdx.x * blockDim.x + threadIdx.x;       // 73728
    if (idx >= Bn * KK * HW1) return;
    int j = idx & 31;
    int i = (idx >> 5) & 63;
    int k = (idx >> 11) % KK;
    int b = idx / (KK * HW1);
    int n = i * W1 + j;

    float oy = off[(((size_t)b * 18 + 2 * k) * HW1) + n];
    float ox = off[(((size_t)b * 18 + 2 * k + 1) * HW1) + n];
    float ys = (float)i + (float)(k / 3 - 1) + oy;
    float xs = (float)j + (float)(k % 3 - 1) + ox;

    float y0f = floorf(ys), x0f = floorf(xs);
    float ly = ys - y0f, lx = xs - x0f;
    int y0 = (int)y0f, x0 = (int)x0f;
    int y1 = y0 + 1, x1 = x0 + 1;

    float vy0 = (y0 >= 0 && y0 < H1) ? 1.f : 0.f;
    float vy1 = (y1 >= 0 && y1 < H1) ? 1.f : 0.f;
    float vx0 = (x0 >= 0 && x0 < W1) ? 1.f : 0.f;
    float vx1 = (x1 >= 0 && x1 < W1) ? 1.f : 0.f;
    int yc0 = iclamp(y0, 0, H1 - 1) * W1, yc1 = iclamp(y1, 0, H1 - 1) * W1;
    int xc0 = iclamp(x0, 0, W1 - 1),      xc1 = iclamp(x1, 0, W1 - 1);

    float w00 = (1.f - ly) * (1.f - lx) * vy0 * vx0;
    float w01 = (1.f - ly) * lx         * vy0 * vx1;
    float w10 = ly         * (1.f - lx) * vy1 * vx0;
    float w11 = ly         * lx         * vy1 * vx1;

    const float* xb = x + (size_t)b * Cc * 4096;
    __bf16* dst = BmT + ((size_t)b * HW1 + n) * CK + k;
    for (int c = 0; c < Cc; ++c) {
        const float* p = xb + (size_t)c * 4096;            // X1 (64,32) view
        float s = w00 * p[yc0 + xc0] + w01 * p[yc0 + xc1]
                + w10 * p[yc1 + xc0] + w11 * p[yc1 + xc1];
        dst[(size_t)c * KK] = (__bf16)s;
    }
}

// ---------------------------------------------------------------------------
// K3a: K-major im2col for the dilated conv (dilation 2, pad 2) on X2 view.
// Thread index ordered (b, n, c, k) so stores are fully coalesced.
// ---------------------------------------------------------------------------
__global__ __launch_bounds__(256)
void dilated_im2col_kernel(const float* __restrict__ x,
                           __bf16* __restrict__ BmT)       // (Bn,2048,1152)
{
    int idx = blockIdx.x * blockDim.x + threadIdx.x;       // 9437184
    if (idx >= Bn * HW1 * CK) return;
    int k = idx % KK;
    int c = (idx / KK) & (Cc - 1);
    int n = (idx / CK) & (HW1 - 1);
    int b = idx / (HW1 * CK);
    int i = n >> 5, j = n & 31;
    int ii = i + 2 * (k / 3) - 2;
    int jj = j + 2 * (k % 3) - 2;
    float v = 0.f;
    if (ii >= 0 && ii < H1 && jj >= 0 && jj < W1) {
        v = x[((size_t)b * Cc + c) * 4096 + 2048 + ii * W1 + jj];  // X2 half
    }
    BmT[idx] = (__bf16)v;
}

// ---------------------------------------------------------------------------
// Conv GEMM: Out[b](128 x 2048) = A(128 x 1152) * B[b](1152 x 2048),
// B given K-major (BmT[n][kk]).  bf16 in, f32 WMMA accum, bf16 out (plain
// (B,128,64,32) layout — this doubles as the "transposed" operand downstream).
// One wave: 16(M) x 64(N), i.e. 1 A fragment feeds 4 WMMAs per K-step.
// All fragment loads are 128-bit.
// ---------------------------------------------------------------------------
__global__ __launch_bounds__(128)
void gemm_conv_kernel(const __bf16* __restrict__ A,     // (128,1152)
                      const __bf16* __restrict__ BmT,   // (Bn,2048,1152)
                      __bf16* __restrict__ Out)         // (Bn,128,2048)
{
    const int wave = threadIdx.x >> 5;
    const int lane = threadIdx.x & 31;
    const int lm = lane & 15, hi = lane >> 4;

    int job = blockIdx.x * 4 + wave;     // 0..1023 (256 jobs/batch: 8 M x 32 Ngrp)
    int b  = job >> 8;
    int t  = job & 255;
    int mt = t >> 5;                     // 0..7
    int ng = t & 31;                     // 0..31
    const int m0 = mt * 16, n0 = ng * 64;

    const __bf16* Bb = BmT + (size_t)b * HW1 * CK;
    v8f acc[4] = {v8f{}, v8f{}, v8f{}, v8f{}};
    for (int ks = 0; ks < CK; ks += 32) {
        const __bf16* Arow = A + (size_t)(m0 + lm) * CK + ks;
        __builtin_prefetch((const void*)(Arow + 32), 0, 0);   // global_prefetch_b8
        // A lane layout: p<8 -> k=hi*8+p ; p>=8 -> k=16+hi*8+(p-8): two b128 runs
        v16bf afrag = cat16(*(const v8bf*)(Arow + hi * 8),
                            *(const v8bf*)(Arow + 16 + hi * 8));
#pragma unroll
        for (int s = 0; s < 4; ++s) {
            // B lane layout: p<8 -> k=hi*8+p ; p>=8 -> k=16+hi*8+(p-8): two b128 runs
            const __bf16* Brow = Bb + (size_t)(n0 + s * 16 + lm) * CK + ks;
            v16bf bfrag = cat16(*(const v8bf*)(Brow + hi * 8),
                                *(const v8bf*)(Brow + 16 + hi * 8));
            acc[s] = __builtin_amdgcn_wmma_f32_16x16x32_bf16(
                         false, afrag, false, bfrag, (short)0, acc[s], false, false);
        }
    }
#pragma unroll
    for (int s = 0; s < 4; ++s) {
#pragma unroll
        for (int r = 0; r < 8; ++r) {
            int m = m0 + r + hi * 8;
            int n = n0 + s * 16 + lm;
            Out[((size_t)b * OC + m) * HW1 + n] = (__bf16)acc[s][r];
        }
    }
}

// ---------------------------------------------------------------------------
// K4: batched matmul, per (b,c): (64x32) @ (32x64) -> (64x64) f32 to d_out.
// X1m[m][k] is K-contiguous; X2m stored plain (i,j)=(n,k) IS the transposed
// operand read K-contiguously.  K=32 -> one WMMA per 16x16 tile.
// ---------------------------------------------------------------------------
__global__ __launch_bounds__(128)
void batched_mm_kernel(const __bf16* __restrict__ X1,   // (Bn,128,64,32) [m][k]
                       const __bf16* __restrict__ X2,   // (Bn,128,64,32) [n][k]
                       float* __restrict__ Out)         // (Bn,128,64,64)
{
    int bc = blockIdx.x;                                // 0..511
    const __bf16* Am = X1 + (size_t)bc * 64 * 32;
    const __bf16* Bm = X2 + (size_t)bc * 64 * 32;
    float* O = Out + (size_t)bc * 64 * 64;

    int wave = threadIdx.x >> 5;
    int lane = threadIdx.x & 31;
    int lm = lane & 15, hi = lane >> 4;

    const __bf16* Arow = Am + (size_t)(wave * 16 + lm) * 32;
    v16bf afrag = cat16(*(const v8bf*)(Arow + hi * 8),
                        *(const v8bf*)(Arow + 16 + hi * 8));
#pragma unroll
    for (int nt = 0; nt < 4; ++nt) {
        const __bf16* Brow = Bm + (size_t)(nt * 16 + lm) * 32;
        v16bf bfrag = cat16(*(const v8bf*)(Brow + hi * 8),
                            *(const v8bf*)(Brow + 16 + hi * 8));
        v8f acc = {};
        acc = __builtin_amdgcn_wmma_f32_16x16x32_bf16(
                  false, afrag, false, bfrag, (short)0, acc, false, false);
#pragma unroll
        for (int r = 0; r < 8; ++r) {
            int m = wave * 16 + r + hi * 8;
            int n = nt * 16 + lm;
            O[(size_t)m * 64 + n] = acc[r];
        }
    }
}

// ---------------------------------------------------------------------------
// Launcher
// ---------------------------------------------------------------------------
extern "C" void kernel_launch(void* const* d_in, const int* in_sizes, int n_in,
                              void* d_out, int out_size, void* d_ws, size_t ws_size,
                              hipStream_t stream) {
    const float* x         = (const float*)d_in[0];   // (4,128,64,64)
    const float* offset_w  = (const float*)d_in[1];   // (18,128,3,3)
    const float* offset_b  = (const float*)d_in[2];   // (18,)
    const float* deform_w  = (const float*)d_in[3];   // (128,128,3,3)
    const float* dilated_w = (const float*)d_in[4];   // (128,128,3,3)
    float* out = (float*)d_out;                       // (4,128,64,64)

    char* ws = (char*)d_ws;
    auto carve = [&](size_t bytes) -> char* {
        char* p = ws;
        ws += (bytes + 255) & ~(size_t)255;
        return p;
    };
    float*  off  = (float*) carve((size_t)Bn * 18 * HW1 * 4);        // 0.59 MB
    __bf16* Adef = (__bf16*)carve((size_t)OC * CK * 2);              // 0.29 MB
    __bf16* Adil = (__bf16*)carve((size_t)OC * CK * 2);              // 0.29 MB
    __bf16* Bdef = (__bf16*)carve((size_t)Bn * HW1 * CK * 2);        // 18.9 MB
    __bf16* Bdil = (__bf16*)carve((size_t)Bn * HW1 * CK * 2);        // 18.9 MB
    __bf16* X1m  = (__bf16*)carve((size_t)Bn * OC * HW1 * 2);        // 2.1 MB
    __bf16* X2m  = (__bf16*)carve((size_t)Bn * OC * HW1 * 2);        // 2.1 MB

    const int nW = OC * CK;   // 147456
    cvt_bf16_kernel<<<(nW + 255) / 256, 256, 0, stream>>>(deform_w,  Adef, nW);
    cvt_bf16_kernel<<<(nW + 255) / 256, 256, 0, stream>>>(dilated_w, Adil, nW);

    const int nOff = Bn * 18 * HW1;                   // 147456
    offset_conv_kernel<<<(nOff + 255) / 256, 256, 0, stream>>>(x, offset_w, offset_b, off);

    const int nSamp = Bn * KK * HW1;                  // 73728
    deform_sample_kernel<<<(nSamp + 255) / 256, 256, 0, stream>>>(x, off, Bdef);

    const int nCol = Bn * HW1 * CK;                   // 9437184
    dilated_im2col_kernel<<<(nCol + 255) / 256, 256, 0, stream>>>(x, Bdil);

    // 1024 wave-jobs per GEMM, 4 waves/WG -> 256 WGs
    gemm_conv_kernel<<<256, 128, 0, stream>>>(Adef, Bdef, X1m);
    gemm_conv_kernel<<<256, 128, 0, stream>>>(Adil, Bdil, X2m);

    batched_mm_kernel<<<512, 128, 0, stream>>>(X1m, X2m, out);
}
